// MambaTasNet_40802189312120
// MI455X (gfx1250) — matrix operand order repaired and checked by reading, not compiled, not run on hardware
//
#include <hip/hip_runtime.h>
#include <hip/hip_bf16.h>

// ---------------- model constants ----------------
constexpr int BSZ = 2;
constexpr int TLEN = 8000;
constexpr int NENC = 256;   // encoder channels
constexpr int KKER = 16;    // encoder kernel
constexpr int STR  = 8;     // encoder stride
constexpr int LFR  = 999;   // frames = (8000-16)/8+1
constexpr int MV   = BSZ * LFR;   // 1998 valid tokens
constexpr int MP   = 2016;        // padded to 63*32 (32-row GEMM tiles)
constexpr int BOT  = 256;
constexpr int DI   = 512;
constexpr int DST  = 16;    // state dim
constexpr int DTR  = 16;
constexpr int NMB  = 8;
constexpr int XDS  = 64;    // padded xproj output stride (48 -> 64)

// ---------------- WMMA types ----------------
typedef __attribute__((ext_vector_type(16))) __bf16          v16bf;
typedef __attribute__((ext_vector_type(8)))  float           v8f;
typedef __attribute__((ext_vector_type(16))) unsigned short  v16u;
typedef __attribute__((ext_vector_type(8)))  unsigned short  v8u;

__device__ __forceinline__ unsigned short f2bf(float f) {
  unsigned int u = __float_as_uint(f);
  unsigned int r = u + 0x7FFFu + ((u >> 16) & 1u);   // RNE
  return (unsigned short)(r >> 16);
}

__device__ __forceinline__ float siluf(float x) {
  return x / (1.0f + __expf(-x));
}

// A fragment (16x32 bf16): lanes 0-15 row M, K{0..7,16..23}; lanes 16-31 K{8..15,24..31}
__device__ __forceinline__ v16bf load_afrag(const unsigned short* row, int lhalf) {
  v8u lo = *(const v8u*)(row + lhalf * 8);
  v8u hi = *(const v8u*)(row + lhalf * 8 + 16);
  v16u f;
  #pragma unroll
  for (int i = 0; i < 8; ++i) { f[i] = lo[i]; f[i + 8] = hi[i]; }
  union { v16u u; v16bf b; } c; c.u = f; return c.b;
}

// B fragment (32x16 bf16): col = lane%16, K{0..7,16..23} lanes 0-15, K{8..15,24..31} lanes 16-31 (W rows are B^T rows)
__device__ __forceinline__ v16bf load_bfrag(const unsigned short* row, int lhalf) {
  v8u lo = *(const v8u*)(row + lhalf * 8);
  v8u hi = *(const v8u*)(row + lhalf * 8 + 16);
  v16u f;
  #pragma unroll
  for (int i = 0; i < 8; ++i) { f[i] = lo[i]; f[i + 8] = hi[i]; }
  union { v16u u; v16bf b; } c; c.u = f; return c.b;
}

// ---------------- small utility kernels ----------------
__global__ void k_zero(float* p, int n) {
  int g = blockIdx.x * blockDim.x + threadIdx.x;
  if (g < n) p[g] = 0.f;
}

__global__ void k_cvt_bf16(unsigned short* dst, const float* src, int n) {
  int g = blockIdx.x * blockDim.x + threadIdx.x;
  if (g < n) dst[g] = f2bf(src[g]);
}

// xproj weights: (NMB*2, 48, 512) f32 -> (NMB*2, 64, 512) bf16, rows 48..63 zero
__global__ void k_cvt_xproj(unsigned short* dst, const float* src) {
  int g = blockIdx.x * blockDim.x + threadIdx.x;
  if (g >= NMB * 2 * XDS * DI) return;
  int kk = g & (DI - 1);
  int rr = (g >> 9) & (XDS - 1);
  int w  = g >> 15;
  float v = (rr < 48) ? src[((size_t)w * 48 + rr) * DI + kk] : 0.f;
  dst[g] = f2bf(v);
}

// ---------------- encoder: conv1d (VALID, stride 8) + relu ----------------
__global__ void k_encoder(const float* __restrict__ mix,
                          const float* __restrict__ enc_w,
                          float* __restrict__ mw) {
  int g = blockIdx.x * blockDim.x + threadIdx.x;
  if (g >= BSZ * NENC * LFR) return;
  int l = g % LFR;
  int n = (g / LFR) % NENC;
  int b = g / (LFR * NENC);
  const float* w = enc_w + n * KKER;
  const float* x = mix + b * TLEN + l * STR;
  float acc = 0.f;
  #pragma unroll
  for (int k = 0; k < KKER; ++k) acc += w[k] * x[k];
  mw[g] = fmaxf(acc, 0.f);
}

// per-batch sum / sumsq over (N, L)
__global__ void k_stats(const float* __restrict__ mw, float* __restrict__ stats) {
  __shared__ float s1[256], s2[256];
  int b = blockIdx.y;
  int idx = blockIdx.x * 256 + threadIdx.x;
  float v = 0.f;
  if (idx < NENC * LFR) v = mw[b * NENC * LFR + idx];
  s1[threadIdx.x] = v;
  s2[threadIdx.x] = v * v;
  __syncthreads();
  for (int s = 128; s > 0; s >>= 1) {
    if (threadIdx.x < s) { s1[threadIdx.x] += s1[threadIdx.x + s]; s2[threadIdx.x] += s2[threadIdx.x + s]; }
    __syncthreads();
  }
  if (threadIdx.x == 0) {
    atomicAdd(&stats[b * 2 + 0], s1[0]);
    atomicAdd(&stats[b * 2 + 1], s2[0]);
  }
}

// groupnorm + transpose (B,N,L) -> token-major bf16 [MP][256]
__global__ void k_gn_pack(const float* __restrict__ mw,
                          const float* __restrict__ stats,
                          const float* __restrict__ gnw,
                          const float* __restrict__ gnb,
                          unsigned short* __restrict__ out) {
  int g = blockIdx.x * blockDim.x + threadIdx.x;
  if (g >= MP * NENC) return;
  int n = g & 255;
  int m = g >> 8;
  unsigned short o = 0;
  if (m < MV) {
    int b = m / LFR, l = m % LFR;
    float cnt = (float)(NENC * LFR);
    float mu = stats[b * 2 + 0] / cnt;
    float var = stats[b * 2 + 1] / cnt - mu * mu;
    float v = mw[(b * NENC + n) * LFR + l];
    o = f2bf((v - mu) * rsqrtf(var + 1e-5f) * gnw[n] + gnb[n]);
  }
  out[g] = o;
}

// ---------------- WMMA GEMM, 32x32 tile per wave (2x2 register blocking)
// C[m][n] = sum_k A[m][k]*W[n][k] (+bias)(+relu)
// A: [MP][Kd] bf16 row-major (tokens), W: [Nn][Kd] bf16 row-major (out,in).
// Kd multiple of 32, Nn multiple of 32; full EXEC, no divergence.
__global__ __launch_bounds__(32)
void k_gemm_bf16(const unsigned short* __restrict__ A,
                 const unsigned short* __restrict__ W,
                 const float* __restrict__ bias,
                 float* __restrict__ C,
                 int Kd, int Nn, int relu) {
  const int lane  = threadIdx.x & 31;
  const int lhalf = lane >> 4;
  const int lmod  = lane & 15;
  const int tN = blockIdx.x;   // 32 cols
  const int tM = blockIdx.y;   // 32 rows

  const unsigned short* arow0 = A + (size_t)(tM * 32 + lmod) * Kd;
  const unsigned short* arow1 = arow0 + (size_t)16 * Kd;
  const unsigned short* wrow0 = W + (size_t)(tN * 32 + lmod) * Kd;
  const unsigned short* wrow1 = wrow0 + (size_t)16 * Kd;

  v8f acc00 = {0.f,0.f,0.f,0.f,0.f,0.f,0.f,0.f};
  v8f acc01 = acc00, acc10 = acc00, acc11 = acc00;

  for (int k0 = 0; k0 < Kd; k0 += 32) {
    v16bf a0 = load_afrag(arow0 + k0, lhalf);
    v16bf a1 = load_afrag(arow1 + k0, lhalf);
    v16bf b0 = load_bfrag(wrow0 + k0, lhalf);
    v16bf b1 = load_bfrag(wrow1 + k0, lhalf);
    acc00 = __builtin_amdgcn_wmma_f32_16x16x32_bf16(false, a0, false, b0, (short)0, acc00, false, false);
    acc01 = __builtin_amdgcn_wmma_f32_16x16x32_bf16(false, a0, false, b1, (short)0, acc01, false, false);
    acc10 = __builtin_amdgcn_wmma_f32_16x16x32_bf16(false, a1, false, b0, (short)0, acc10, false, false);
    acc11 = __builtin_amdgcn_wmma_f32_16x16x32_bf16(false, a1, false, b1, (short)0, acc11, false, false);
  }

  const int cn0 = tN * 32 + lmod;
  const int cm  = tM * 32 + lhalf * 8;   // D: VGPR r -> row r (+8 for hi half)
  float bv0 = bias ? bias[cn0] : 0.f;
  float bv1 = bias ? bias[cn0 + 16] : 0.f;
  const v8f* accs[4] = { &acc00, &acc01, &acc10, &acc11 };
  #pragma unroll
  for (int ti = 0; ti < 2; ++ti) {
    #pragma unroll
    for (int tj = 0; tj < 2; ++tj) {
      const v8f& a = *accs[ti * 2 + tj];
      float bv = tj ? bv1 : bv0;
      int cn = cn0 + tj * 16;
      #pragma unroll
      for (int r = 0; r < 8; ++r) {
        float v = a[r] + bv;
        if (relu) v = fmaxf(v, 0.f);
        C[(size_t)(cm + ti * 16 + r) * Nn + cn] = v;
      }
    }
  }
}

// ---------------- rmsnorm over 256 features, emit bf16 fwd (+optional reversed) --
__global__ __launch_bounds__(256)
void k_rmsnorm(const float* __restrict__ x,
               const float* __restrict__ w,
               unsigned short* __restrict__ outf,
               unsigned short* __restrict__ outr,
               int do_rev) {
  __shared__ float red[256];
  int m = blockIdx.x;
  int c = threadIdx.x;
  float v = (m < MV) ? x[(size_t)m * BOT + c] : 0.f;
  red[c] = v * v;
  __syncthreads();
  for (int s = 128; s > 0; s >>= 1) {
    if (c < s) red[c] += red[c + s];
    __syncthreads();
  }
  float scale = rsqrtf(red[0] / (float)BOT + 1e-5f);
  unsigned short o = f2bf(v * w[c] * scale);
  outf[(size_t)m * BOT + c] = o;
  if (do_rev) {
    int mr = m;
    if (m < MV) { int b = m / LFR, l = m % LFR; mr = b * LFR + (LFR - 1 - l); }
    outr[(size_t)mr * BOT + c] = o;
  }
}

// ---------------- depthwise causal conv(4) + bias + SiLU ----------------
__global__ void k_conv_silu(const float* __restrict__ xz,
                            const float* __restrict__ cw,
                            const float* __restrict__ cb,
                            float* __restrict__ xc,
                            unsigned short* __restrict__ xcbf) {
  int g = blockIdx.x * blockDim.x + threadIdx.x;
  if (g >= MP * DI) return;
  int d = g & (DI - 1);
  int m = g >> 9;
  float outv = 0.f;
  if (m < MV) {
    int b = m / LFR, l = m % LFR;
    float acc = cb[d];
    #pragma unroll
    for (int j = 0; j < 4; ++j) {
      int li = l - 3 + j;
      if (li >= 0) acc += cw[d * 4 + j] * xz[(size_t)(b * LFR + li) * (2 * DI) + d];
    }
    outv = siluf(acc);
  }
  xc[g] = outv;
  xcbf[g] = f2bf(outv);
}

// ---------------- dt = softplus(xdbl[:, :16] @ dt_w^T + dt_b) ----------------
__global__ void k_dt(const float* __restrict__ xdbl,
                     const float* __restrict__ dtw,
                     const float* __restrict__ dtb,
                     float* __restrict__ dt) {
  int g = blockIdx.x * blockDim.x + threadIdx.x;
  if (g >= MP * DI) return;
  int d = g & (DI - 1);
  int m = g >> 9;
  float acc = dtb[d];
  const float* row = xdbl + (size_t)m * XDS;
  #pragma unroll
  for (int r = 0; r < DTR; ++r) acc += row[r] * dtw[d * DTR + r];
  dt[g] = (acc > 20.f) ? acc : log1pf(__expf(acc));
}

// ---------------- selective scan: lane per (b,d,s); shfl reduce over s --------
__global__ __launch_bounds__(256)
void k_scan(const float* __restrict__ dt,
            const float* __restrict__ xc,
            const float* __restrict__ xdbl,
            const float* __restrict__ Alog,
            const float* __restrict__ Dp,
            float* __restrict__ ys) {
  int g = blockIdx.x * blockDim.x + threadIdx.x;   // exactly 2*512*16 = 16384
  int s = g & 15;
  int d = (g >> 4) & (DI - 1);
  int b = g >> 13;
  float A = -__expf(Alog[d * DST + s]);
  float Dv = Dp[d];
  float h = 0.f;
  int base = b * LFR;
  for (int l = 0; l < LFR; ++l) {
    int m = base + l;
    float dtv = dt[(size_t)m * DI + d];
    float xv  = xc[(size_t)m * DI + d];
    float Bv  = xdbl[(size_t)m * XDS + 16 + s];
    float Cv  = xdbl[(size_t)m * XDS + 32 + s];
    h = __expf(dtv * A) * h + (dtv * xv) * Bv;
    float p = h * Cv;
    p += __shfl_xor(p, 1);
    p += __shfl_xor(p, 2);
    p += __shfl_xor(p, 4);
    p += __shfl_xor(p, 8);
    if (s == 0) ys[(size_t)m * DI + d] = p + xv * Dv;
  }
}

// ---------------- gate: y * silu(z), pack bf16 ----------------
__global__ void k_gate(const float* __restrict__ ys,
                       const float* __restrict__ xz,
                       unsigned short* __restrict__ ygbf) {
  int g = blockIdx.x * blockDim.x + threadIdx.x;
  if (g >= MP * DI) return;
  int d = g & (DI - 1);
  int m = g >> 9;
  unsigned short o = 0;
  if (m < MV) {
    float z = xz[(size_t)m * (2 * DI) + DI + d];
    o = f2bf(ys[g] * siluf(z));
  }
  ygbf[g] = o;
}

// ---------------- residual: x += yf + flip(yb) ----------------
__global__ void k_residual(float* __restrict__ x,
                           const float* __restrict__ yd0,
                           const float* __restrict__ yd1) {
  int g = blockIdx.x * blockDim.x + threadIdx.x;
  if (g >= MV * BOT) return;
  int c = g & 255;
  int m = g >> 8;
  int b = m / LFR, l = m % LFR;
  int mr = b * LFR + (LFR - 1 - l);
  x[g] += yd0[g] + yd1[(size_t)mr * BOT + c];
}

// ---------------- masked = mw * relu-mask (token-major out) ----------------
__global__ void k_apply_mask(const float* __restrict__ mw,
                             const float* __restrict__ masks,
                             float* __restrict__ masked) {
  int g = blockIdx.x * blockDim.x + threadIdx.x;
  if (g >= MP * NENC) return;
  int n = g & 255;
  int m = g >> 8;
  float v = 0.f;
  if (m < MV) {
    int b = m / LFR, l = m % LFR;
    v = mw[(b * NENC + n) * LFR + l] * masks[g];
  }
  masked[g] = v;
}

// ---------------- decoder: transposed conv (lhs_dilation=8, pad 15) ----------
__global__ void k_decoder(const float* __restrict__ masked,
                          const float* __restrict__ dec_w,
                          float* __restrict__ out) {
  int g = blockIdx.x * blockDim.x + threadIdx.x;
  if (g >= BSZ * TLEN) return;
  int t = g % TLEN;
  int b = g / TLEN;
  float acc = 0.f;
  #pragma unroll
  for (int k = 0; k < KKER; ++k) {
    int p = t + k - (KKER - 1);
    if (p >= 0 && p < (LFR - 1) * STR + 1 && (p & 7) == 0) {
      int f = p >> 3;
      const float* mt = masked + (size_t)(b * LFR + f) * NENC;
      int kk = KKER - 1 - k;
      float a = 0.f;
      for (int n = 0; n < NENC; ++n) a += mt[n] * dec_w[n * KKER + kk];
      acc += a;
    }
  }
  out[g] = acc;
}

// ---------------- host launch ----------------
static inline size_t alignup(size_t x) { return (x + 255) & ~(size_t)255; }

extern "C" void kernel_launch(void* const* d_in, const int* in_sizes, int n_in,
                              void* d_out, int out_size, void* d_ws, size_t ws_size,
                              hipStream_t stream) {
  (void)in_sizes; (void)n_in; (void)out_size; (void)ws_size;

  const float* mixture = (const float*)d_in[0];
  const float* enc_w   = (const float*)d_in[1];
  const float* dec_w   = (const float*)d_in[2];
  const float* gn_w    = (const float*)d_in[3];
  const float* gn_b    = (const float*)d_in[4];
  const float* bot_w   = (const float*)d_in[5];
  const float* bot_b   = (const float*)d_in[6];
  const float* blk_nw  = (const float*)d_in[7];
  const float* in_w    = (const float*)d_in[8];
  const float* conv_w  = (const float*)d_in[9];
  const float* conv_b  = (const float*)d_in[10];
  const float* xproj_w = (const float*)d_in[11];
  const float* dt_w    = (const float*)d_in[12];
  const float* dt_b    = (const float*)d_in[13];
  const float* A_log   = (const float*)d_in[14];
  const float* Dmat    = (const float*)d_in[15];
  const float* out_w   = (const float*)d_in[16];
  const float* normf_w = (const float*)d_in[17];
  const float* mask_w  = (const float*)d_in[18];
  const float* mask_b  = (const float*)d_in[19];
  float* y_out = (float*)d_out;

  char* ws = (char*)d_ws;
  size_t off = 0;
  auto take = [&](size_t bytes) { size_t o = off; off = alignup(off + bytes); return o; };

  float*          MW    = (float*)(ws + take((size_t)BSZ * NENC * LFR * 4));
  float*          STATS = (float*)(ws + take(16));
  unsigned short* A0BF  = (unsigned short*)(ws + take((size_t)MP * NENC * 2));
  float*          XRES  = (float*)(ws + take((size_t)MP * BOT * 4));
  unsigned short* XNBF  = (unsigned short*)(ws + take((size_t)MP * BOT * 2));
  unsigned short* XNRBF = (unsigned short*)(ws + take((size_t)MP * BOT * 2));
  float*          XZ    = (float*)(ws + take((size_t)MP * 2 * DI * 4));
  float*          XC    = (float*)(ws + take((size_t)MP * DI * 4));
  unsigned short* XCBF  = (unsigned short*)(ws + take((size_t)MP * DI * 2));
  float*          XDBL  = (float*)(ws + take((size_t)MP * XDS * 4));
  float*          DT    = (float*)(ws + take((size_t)MP * DI * 4));
  float*          YS    = (float*)(ws + take((size_t)MP * DI * 4));
  unsigned short* YGBF  = (unsigned short*)(ws + take((size_t)MP * DI * 2));
  float*          YD0   = (float*)(ws + take((size_t)MP * BOT * 4));
  float*          YD1   = (float*)(ws + take((size_t)MP * BOT * 4));
  float*          MASKS = (float*)(ws + take((size_t)MP * NENC * 4));
  float*          MASKED= (float*)(ws + take((size_t)MP * NENC * 4));
  unsigned short* WBOT  = (unsigned short*)(ws + take((size_t)BOT * NENC * 2));
  unsigned short* WIN   = (unsigned short*)(ws + take((size_t)NMB * 2 * 2 * DI * BOT * 2));
  unsigned short* WXP   = (unsigned short*)(ws + take((size_t)NMB * 2 * XDS * DI * 2));
  unsigned short* WOUT  = (unsigned short*)(ws + take((size_t)NMB * 2 * BOT * DI * 2));
  unsigned short* WMASK = (unsigned short*)(ws + take((size_t)NENC * BOT * 2));

  auto nblk = [](long n) { return (unsigned)((n + 255) / 256); };

  // ---- weight conversion to bf16 (once per call; deterministic) ----
  k_cvt_bf16<<<nblk((long)BOT * NENC), 256, 0, stream>>>(WBOT, bot_w, BOT * NENC);
  k_cvt_bf16<<<nblk((long)NMB * 2 * 2 * DI * BOT), 256, 0, stream>>>(WIN, in_w, NMB * 2 * 2 * DI * BOT);
  k_cvt_xproj<<<nblk((long)NMB * 2 * XDS * DI), 256, 0, stream>>>(WXP, xproj_w);
  k_cvt_bf16<<<nblk((long)NMB * 2 * BOT * DI), 256, 0, stream>>>(WOUT, out_w, NMB * 2 * BOT * DI);
  k_cvt_bf16<<<nblk((long)NENC * BOT), 256, 0, stream>>>(WMASK, mask_w, NENC * BOT);

  // ---- encoder + groupnorm + bottleneck ----
  k_zero<<<1, 32, 0, stream>>>(STATS, 4);
  k_encoder<<<nblk((long)BSZ * NENC * LFR), 256, 0, stream>>>(mixture, enc_w, MW);
  k_stats<<<dim3(nblk((long)NENC * LFR), BSZ), 256, 0, stream>>>(MW, STATS);
  k_gn_pack<<<nblk((long)MP * NENC), 256, 0, stream>>>(MW, STATS, gn_w, gn_b, A0BF);
  k_gemm_bf16<<<dim3(BOT / 32, MP / 32), 32, 0, stream>>>(A0BF, WBOT, bot_b, XRES, NENC, BOT, 0);

  // ---- mamba blocks ----
  for (int i = 0; i < NMB; ++i) {
    k_rmsnorm<<<MP, 256, 0, stream>>>(XRES, blk_nw + i * BOT, XNBF, XNRBF, 1);
    for (int dir = 0; dir < 2; ++dir) {
      int w = i * 2 + dir;
      const unsigned short* tok = dir ? XNRBF : XNBF;
      // in_proj: [MP,256] x [1024,256] -> XZ [MP,1024]
      k_gemm_bf16<<<dim3((2 * DI) / 32, MP / 32), 32, 0, stream>>>(
          tok, WIN + (size_t)w * 2 * DI * BOT, nullptr, XZ, BOT, 2 * DI, 0);
      k_conv_silu<<<nblk((long)MP * DI), 256, 0, stream>>>(
          XZ, conv_w + (size_t)w * DI * 4, conv_b + (size_t)w * DI, XC, XCBF);
      // xproj: [MP,512] x [64,512] -> XDBL [MP,64] (rows 48..63 zero-padded)
      k_gemm_bf16<<<dim3(XDS / 32, MP / 32), 32, 0, stream>>>(
          XCBF, WXP + (size_t)w * XDS * DI, nullptr, XDBL, DI, XDS, 0);
      k_dt<<<nblk((long)MP * DI), 256, 0, stream>>>(
          XDBL, dt_w + (size_t)w * DI * DTR, dt_b + (size_t)w * DI, DT);
      k_scan<<<(BSZ * DI * DST) / 256, 256, 0, stream>>>(
          DT, XC, XDBL, A_log + (size_t)w * DI * DST, Dmat + (size_t)w * DI, YS);
      k_gate<<<nblk((long)MP * DI), 256, 0, stream>>>(YS, XZ, YGBF);
      // out_proj: [MP,512] x [256,512] -> YD [MP,256]
      k_gemm_bf16<<<dim3(BOT / 32, MP / 32), 32, 0, stream>>>(
          YGBF, WOUT + (size_t)w * BOT * DI, nullptr, dir ? YD1 : YD0, DI, BOT, 0);
    }
    k_residual<<<nblk((long)MV * BOT), 256, 0, stream>>>(XRES, YD0, YD1);
  }

  // ---- final norm + mask + decode ----
  k_rmsnorm<<<MP, 256, 0, stream>>>(XRES, normf_w, XNBF, XNRBF, 0);
  k_gemm_bf16<<<dim3(NENC / 32, MP / 32), 32, 0, stream>>>(
      XNBF, WMASK, mask_b, MASKS, BOT, NENC, 1);
  k_apply_mask<<<nblk((long)MP * NENC), 256, 0, stream>>>(MW, MASKS, MASKED);
  k_decoder<<<nblk((long)BSZ * TLEN), 256, 0, stream>>>(MASKED, dec_w, y_out);
}
